// GATLayer_38474317037826
// MI455X (gfx1250) — compile-verified, matrix operand order checked by reading, not run on hardware
//
#include <hip/hip_runtime.h>
#include <hip/hip_bf16.h>
#include <math.h>

// ---------------------------------------------------------------------------
// GAT layer: z = x@W (WMMA bf16), per-vertex attention scores, segment
// softmax over dst via atomics, scatter-aggregate, ELU.
// V=100000, E=1.6M, Fin=128, H=4, Cout=32 (H*Cout=128).
// ---------------------------------------------------------------------------

typedef __attribute__((ext_vector_type(16))) __bf16 v16bf;
typedef __attribute__((ext_vector_type(8)))  float  v8f;

static constexpr int kFin  = 128;
static constexpr int kNcol = 128;   // H*Cout
static constexpr int kH    = 4;
static constexpr int kCout = 32;
static constexpr float kNegInit = -1e9f;

// Monotone order-preserving float<->uint mapping for atomic max on floats.
__device__ __forceinline__ unsigned ford(float f) {
  unsigned u = __float_as_uint(f);
  return (u & 0x80000000u) ? ~u : (u | 0x80000000u);
}
__device__ __forceinline__ float unford(unsigned o) {
  return (o & 0x80000000u) ? __uint_as_float(o & 0x7fffffffu)
                           : __uint_as_float(~o);
}

// ---------------------------------------------------------------------------
// Pack W (f32, [128,128] row-major K x N) into bf16 B-fragments laid out so
// that each lane of each (colTile, ktile) loads one contiguous v16bf.
// ISA 16-bit B-layout: lane L(0-15): N=colTile*16+L, elems j<8 -> K=kbase+j,
// j>=8 -> K=16+kbase+(j-8), kbase=(lane>>4)*8.
// Total: 8 colTiles * 4 ktiles * 32 lanes * 16 elems = 16384 bf16 (32 KB).
// ---------------------------------------------------------------------------
__global__ void pack_w_kernel(const float* __restrict__ W,
                              __bf16* __restrict__ Wpk) {
  int t = blockIdx.x * blockDim.x + threadIdx.x;
  if (t >= 8 * 4 * 32 * 16) return;
  int j       = t & 15;
  int lane    = (t >> 4) & 31;
  int kt      = (t >> 9) & 3;
  int colTile = t >> 11;
  int kbase   = (lane >> 4) * 8;
  int k       = kt * 32 + (j < 8 ? (kbase + j) : (16 + kbase + (j - 8)));
  int col     = colTile * 16 + (lane & 15);
  Wpk[t] = (__bf16)W[k * kNcol + col];
}

// ---------------------------------------------------------------------------
// z = x @ W via v_wmma_f32_16x16x32_bf16.
// Block = 256 threads = 8 waves; block handles 16 rows, wave w handles
// columns [16w, 16w+16). 4 K-steps of 32.
// ---------------------------------------------------------------------------
__global__ void __launch_bounds__(256)
wmma_gemm_kernel(const float* __restrict__ x, const __bf16* __restrict__ Wpk,
                 float* __restrict__ z, int V) {
  const int wave  = threadIdx.x >> 5;
  const int lane  = threadIdx.x & 31;
  const int row0  = blockIdx.x * 16;
  if (row0 >= V) return;
  const int rowA  = row0 + (lane & 15);
  const int kbase = (lane >> 4) * 8;

  v8f acc = {};
#pragma unroll
  for (int kt = 0; kt < 4; ++kt) {
    const int k0 = kt * 32;
    // A fragment: row rowA, K = k0 + {kbase..kbase+7, 16+kbase..16+kbase+7}
    const float* xr = x + (size_t)rowA * kFin + k0 + kbase;
    float4 a0 = *(const float4*)(xr);
    float4 a1 = *(const float4*)(xr + 4);
    float4 a2 = *(const float4*)(xr + 16);
    float4 a3 = *(const float4*)(xr + 20);
    v16bf a;
    a[0]  = (__bf16)a0.x; a[1]  = (__bf16)a0.y; a[2]  = (__bf16)a0.z; a[3]  = (__bf16)a0.w;
    a[4]  = (__bf16)a1.x; a[5]  = (__bf16)a1.y; a[6]  = (__bf16)a1.z; a[7]  = (__bf16)a1.w;
    a[8]  = (__bf16)a2.x; a[9]  = (__bf16)a2.y; a[10] = (__bf16)a2.z; a[11] = (__bf16)a2.w;
    a[12] = (__bf16)a3.x; a[13] = (__bf16)a3.y; a[14] = (__bf16)a3.z; a[15] = (__bf16)a3.w;
    // B fragment: one contiguous 32B load from the packed layout.
    v16bf b = *(const v16bf*)(Wpk + ((size_t)(wave * 4 + kt) * 32 + lane) * 16);
    acc = __builtin_amdgcn_wmma_f32_16x16x32_bf16(
        /*neg_a=*/false, a, /*neg_b=*/false, b,
        /*c_mod=*/(short)0, acc, /*reuse_a=*/false, /*reuse_b=*/false);
  }

  // C/D layout: VGPR r, lanes 0-15 -> M=r, lanes 16-31 -> M=8+r; N=lane&15.
  const int col   = wave * 16 + (lane & 15);
  const int rbase = row0 + ((lane < 16) ? 0 : 8);
#pragma unroll
  for (int r = 0; r < 8; ++r) {
    int rr = rbase + r;
    if (rr < V) z[(size_t)rr * kNcol + col] = acc[r];
  }
}

// ---------------------------------------------------------------------------
// Per-vertex attention half-scores: s_src[v,h]=dot(z[v,h,:],a_src[h,:]), etc.
// ---------------------------------------------------------------------------
__global__ void scores_kernel(const float* __restrict__ z,
                              const float* __restrict__ a_src,
                              const float* __restrict__ a_dst,
                              float* __restrict__ s_src,
                              float* __restrict__ s_dst, int VH) {
  int t = blockIdx.x * blockDim.x + threadIdx.x;
  if (t >= VH) return;
  int v = t >> 2, h = t & 3;
  const float* zr = z + (size_t)v * kNcol + h * kCout;
  const float* as = a_src + h * kCout;
  const float* ad = a_dst + h * kCout;
  float ss = 0.f, sd = 0.f;
#pragma unroll 8
  for (int c = 0; c < kCout; ++c) {
    float zv = zr[c];
    ss = fmaf(zv, as[c], ss);
    sd = fmaf(zv, ad[c], sd);
  }
  s_src[t] = ss;
  s_dst[t] = sd;
}

// ---------------------------------------------------------------------------
// Initialize segment-max (to ord(-1e9)), denominators (0) and output acc (0).
// ---------------------------------------------------------------------------
__global__ void init_kernel(unsigned* __restrict__ maxord,
                            float* __restrict__ denom,
                            float* __restrict__ out, int VH, int VC) {
  int t = blockIdx.x * blockDim.x + threadIdx.x;
  if (t < VH) {
    maxord[t] = ford(kNegInit);
    denom[t]  = 0.f;
  }
  if (t < VC) out[t] = 0.f;
}

__device__ __forceinline__ float edge_logit(const float* s_src,
                                            const float* s_dst, int src,
                                            int dst, int h) {
  float v = s_src[src * kH + h] + s_dst[dst * kH + h];
  return (v >= 0.f) ? v : 0.2f * v;  // LeakyReLU(0.2)
}

// Pass 1: segment max over dst (atomic max via ordered-uint trick).
__global__ void edge_max_kernel(const int* __restrict__ ei,
                                const float* __restrict__ s_src,
                                const float* __restrict__ s_dst,
                                unsigned* __restrict__ maxord, int E, int EH) {
  int t = blockIdx.x * blockDim.x + threadIdx.x;
  if (t >= EH) return;
  int e = t >> 2, h = t & 3;
  int src = ei[e], dst = ei[E + e];
  float v = edge_logit(s_src, s_dst, src, dst, h);
  atomicMax(&maxord[dst * kH + h], ford(v));
}

// Pass 2: denom[dst,h] += exp(e - max[dst,h]).
__global__ void edge_denom_kernel(const int* __restrict__ ei,
                                  const float* __restrict__ s_src,
                                  const float* __restrict__ s_dst,
                                  const unsigned* __restrict__ maxord,
                                  float* __restrict__ denom, int E, int EH) {
  int t = blockIdx.x * blockDim.x + threadIdx.x;
  if (t >= EH) return;
  int e = t >> 2, h = t & 3;
  int src = ei[e], dst = ei[E + e];
  float v = edge_logit(s_src, s_dst, src, dst, h);
  float m = unford(maxord[dst * kH + h]);
  atomicAdd(&denom[dst * kH + h], __expf(v - m));
}

// Pass 3: out[dst, j] += alpha[e, h(j)] * z[src, j], one thread per (edge, j).
__global__ void edge_agg_kernel(const int* __restrict__ ei,
                                const float* __restrict__ s_src,
                                const float* __restrict__ s_dst,
                                const unsigned* __restrict__ maxord,
                                const float* __restrict__ denom,
                                const float* __restrict__ z,
                                float* __restrict__ out, int E, long long EJ) {
  long long t = (long long)blockIdx.x * blockDim.x + threadIdx.x;
  if (t >= EJ) return;
  int e = (int)(t >> 7);
  int j = (int)(t & 127);
  int h = j >> 5;
  int src = ei[e], dst = ei[E + e];
  float v = edge_logit(s_src, s_dst, src, dst, h);
  float m = unford(maxord[dst * kH + h]);
  float alpha = __expf(v - m) / (denom[dst * kH + h] + 1e-9f);
  atomicAdd(&out[(size_t)dst * kNcol + j], z[(size_t)src * kNcol + j] * alpha);
}

// Final ELU in place on the aggregate.
__global__ void elu_kernel(float* __restrict__ out, int VC) {
  int t = blockIdx.x * blockDim.x + threadIdx.x;
  if (t >= VC) return;
  float v = out[t];
  out[t] = (v > 0.f) ? v : expm1f(v);
}

// ---------------------------------------------------------------------------
extern "C" void kernel_launch(void* const* d_in, const int* in_sizes, int n_in,
                              void* d_out, int out_size, void* d_ws,
                              size_t ws_size, hipStream_t stream) {
  const float* x     = (const float*)d_in[0];
  const int*   ei    = (const int*)d_in[1];
  const float* W     = (const float*)d_in[2];
  const float* a_src = (const float*)d_in[3];
  const float* a_dst = (const float*)d_in[4];
  float*       out   = (float*)d_out;

  const int V = in_sizes[0] / kFin;
  const int E = in_sizes[1] / 2;

  // Workspace layout (256B-aligned slabs).
  char* ws = (char*)d_ws;
  size_t off = 0;
  auto alloc = [&](size_t bytes) {
    char* p = ws + off;
    off += (bytes + 255) & ~(size_t)255;
    return p;
  };
  __bf16*   Wpk    = (__bf16*)  alloc(8 * 4 * 32 * 16 * sizeof(__bf16));
  float*    z      = (float*)   alloc((size_t)V * kNcol * sizeof(float));
  float*    s_src  = (float*)   alloc((size_t)V * kH * sizeof(float));
  float*    s_dst  = (float*)   alloc((size_t)V * kH * sizeof(float));
  unsigned* maxord = (unsigned*)alloc((size_t)V * kH * sizeof(unsigned));
  float*    denom  = (float*)   alloc((size_t)V * kH * sizeof(float));

  const int threads = 256;
  const int VH = V * kH;
  const int VC = V * kNcol;
  const int EH = E * kH;
  const long long EJ = (long long)E * kNcol;

  pack_w_kernel<<<(8 * 4 * 32 * 16 + threads - 1) / threads, threads, 0,
                  stream>>>(W, Wpk);
  wmma_gemm_kernel<<<(V + 15) / 16, threads, 0, stream>>>(x, Wpk, z, V);
  scores_kernel<<<(VH + threads - 1) / threads, threads, 0, stream>>>(
      z, a_src, a_dst, s_src, s_dst, VH);
  init_kernel<<<(VC + threads - 1) / threads, threads, 0, stream>>>(
      maxord, denom, out, VH, VC);
  edge_max_kernel<<<(EH + threads - 1) / threads, threads, 0, stream>>>(
      ei, s_src, s_dst, maxord, E, EH);
  edge_denom_kernel<<<(EH + threads - 1) / threads, threads, 0, stream>>>(
      ei, s_src, s_dst, maxord, denom, E, EH);
  edge_agg_kernel<<<(unsigned)((EJ + threads - 1) / threads), threads, 0,
                    stream>>>(ei, s_src, s_dst, maxord, denom, z, out, E, EJ);
  elu_kernel<<<(VC + threads - 1) / threads, threads, 0, stream>>>(out, VC);
}
